// Attention_49907519979595
// MI455X (gfx1250) — matrix operand order repaired and checked by reading, not compiled, not run on hardware
//
#include <hip/hip_runtime.h>

#define H 256
#define S 4096
#define BB 64
#define STILE 128      // s-rows per block
#define NCH 64         // N-chunk of Wr staged in LDS
#define RPAD 264       // padded LDS row stride (bf16 elems) -> avoids bank conflicts

typedef __attribute__((ext_vector_type(16))) __bf16 v16bf;
typedef __attribute__((ext_vector_type(8)))  __bf16 v8bf;
typedef __attribute__((ext_vector_type(8)))  float  v8f;

// round-to-nearest-even f32 -> bf16 bits (manual, avoids any __bf16 cvt quirks)
__device__ __forceinline__ unsigned short bf_bits(float f) {
    unsigned u = __builtin_bit_cast(unsigned, f);
    unsigned lsb = (u >> 16) & 1u;
    u += 0x7fffu + lsb;
    return (unsigned short)(u >> 16);
}
__device__ __forceinline__ unsigned pack_bf2(float a, float b) {
    return (unsigned)bf_bits(a) | ((unsigned)bf_bits(b) << 16);
}

// branchless tanh: copysign(1 - 2/(exp(2|x|)+1), x); saturates for large |x|
__device__ __forceinline__ float fast_tanh(float x) {
    float ax2 = fabsf(x);
    ax2 = ax2 + ax2;
    float e = __expf(ax2);
    float t = 1.0f - 2.0f / (e + 1.0f);
    return copysignf(t, x);
}

// async global -> LDS copy of 16 bytes (GV mode), tracked by ASYNCcnt
__device__ __forceinline__ void async_copy_b128(unsigned lds_byte_off,
                                                const void* gaddr) {
    asm volatile("global_load_async_to_lds_b128 %0, %1, off"
                 :: "v"(lds_byte_off), "v"(gaddr) : "memory");
}
__device__ __forceinline__ void wait_async0() {
    asm volatile("s_wait_asynccnt 0x0" ::: "memory");
}

// ---- kernel 0: Wr f32 -> bf16 -------------------------------------------
__global__ void wr_to_bf16_kernel(const float* __restrict__ Wr,
                                  unsigned short* __restrict__ WrBf) {
    int i = blockIdx.x * blockDim.x + threadIdx.x;   // 0 .. H*H-1
    WrBf[i] = bf_bits(Wr[i]);
}

// ---- kernel 1: q = query @ Wq.T + bq ------------------------------------
__global__ void qproj_kernel(const float* __restrict__ query,
                             const float* __restrict__ Wq,
                             const float* __restrict__ bq,
                             float* __restrict__ q) {
    __shared__ float x[H];
    const int b = blockIdx.x, h = threadIdx.x;
    x[h] = query[b * H + h];
    __syncthreads();
    const float* wrow = Wq + (size_t)h * H;
    float acc = bq[h];
    for (int j = 0; j < H; ++j) acc = __builtin_fmaf(wrow[j], x[j], acc);
    q[b * H + h] = acc;
}

// ---- kernel 2: logits[b,s] = V . tanh(q_b + ref[b,s,:] @ Wr.T + br) -----
__global__ __launch_bounds__(256) void logits_kernel(
    const float* __restrict__ ref, const unsigned short* __restrict__ WrBf,
    const float* __restrict__ qArr, const float* __restrict__ br,
    const float* __restrict__ V, float* __restrict__ logits)
{
    __shared__ __bf16 refT[STILE * RPAD];   // 67,584 B
    __shared__ __bf16 wrT[NCH * RPAD];      // 33,792 B
    __shared__ float  qb[H];
    __shared__ float  Vs[H];

    const int b   = blockIdx.y;
    const int s0  = blockIdx.x * STILE;
    const int tid = threadIdx.x;

    qb[tid] = qArr[b * H + tid] + br[tid];
    Vs[tid] = V[tid];

    // stage ref tile, convert f32 -> bf16 (128 rows x 256 cols), 4-deep batched
    const float4* refv = (const float4*)(ref + ((size_t)b * S + s0) * H);
    for (int i = 0; i < 8; ++i) {
        float4 tmp[4];
        #pragma unroll
        for (int u = 0; u < 4; ++u) tmp[u] = refv[(i * 4 + u) * 256 + tid];
        #pragma unroll
        for (int u = 0; u < 4; ++u) {
            int idx = (i * 4 + u) * 256 + tid;          // float4 index
            int row = idx >> 6;                         // 64 float4 per row
            int col = (idx & 63) << 2;
            unsigned* dst = (unsigned*)&refT[row * RPAD + col];
            dst[0] = pack_bf2(tmp[u].x, tmp[u].y);
            dst[1] = pack_bf2(tmp[u].z, tmp[u].w);
        }
    }
    __syncthreads();

    const int lane  = tid & 31;
    const int l16   = lane & 15;
    const int half  = lane >> 4;
    const int mbase = (tid >> 5) * 16;                  // wave's 16-row tile

    const __bf16* arow = &refT[(mbase + l16) * RPAD];

    float logacc[8] = {0.f, 0.f, 0.f, 0.f, 0.f, 0.f, 0.f, 0.f};

    for (int nc = 0; nc < H / NCH; ++nc) {
        // stage Wr rows [nc*64, nc*64+64) via async global->LDS (bf16, no cvt)
        const uint4* src = (const uint4*)(WrBf + (size_t)nc * NCH * H);
        #pragma unroll
        for (int i = 0; i < (NCH * H / 8) / 256; ++i) { // 8 async b128 per thread
            int idx = i * 256 + tid;                    // uint4 index, 32 per row
            int row = idx >> 5;
            int col = idx & 31;
            unsigned ldsb =
                (unsigned)(unsigned long long)((const char*)&wrT[row * RPAD] + col * 16);
            async_copy_b128(ldsb, (const void*)(src + idx));
        }
        wait_async0();
        __syncthreads();

        #pragma unroll
        for (int nt = 0; nt < 4; ++nt) {
            // B fragment: column n = lane%16, K split by lane half
            const __bf16* brow = &wrT[(nt * 16 + l16) * RPAD + half * 16];
            v8f acc = {};
            #pragma unroll
            for (int kt = 0; kt < 8; ++kt) {
                int kb = kt * 32;
                // A: lanes 0-15 hold K{0..15}, lanes 16-31 K{16..31}
                v8bf a0 = *(const v8bf*)&arow[kb + half * 16];
                v8bf a1 = *(const v8bf*)&arow[kb + half * 16 + 8];
                v16bf av = __builtin_shufflevector(a0, a1,
                    0,1,2,3,4,5,6,7,8,9,10,11,12,13,14,15);
                v8bf b0 = *(const v8bf*)&brow[kb];
                v8bf b1 = *(const v8bf*)&brow[kb + 8];
                v16bf bv = __builtin_shufflevector(b0, b1,
                    0,1,2,3,4,5,6,7,8,9,10,11,12,13,14,15);
                acc = __builtin_amdgcn_wmma_f32_16x16x32_bf16(
                          false, av, false, bv, (short)0, acc, false, false);
            }
            // C/D layout: VGPR j -> row (lane<16 ? j : j+8), col = lane%16
            int n = nc * NCH + nt * 16 + l16;
            float qn = qb[n], vn = Vs[n];
            #pragma unroll
            for (int j = 0; j < 8; ++j)
                logacc[j] = __builtin_fmaf(fast_tanh(qn + acc[j]), vn, logacc[j]);
        }
        __syncthreads();
    }

    // sum columns: reduce within each 16-lane half-group
    #pragma unroll
    for (int j = 0; j < 8; ++j) {
        float v = logacc[j];
        v += __shfl_xor(v, 1, 16);
        v += __shfl_xor(v, 2, 16);
        v += __shfl_xor(v, 4, 16);
        v += __shfl_xor(v, 8, 16);
        logacc[j] = v;
    }
    if (l16 == 0) {
        int sbase = s0 + mbase + half * 8;
        #pragma unroll
        for (int j = 0; j < 8; ++j)
            logits[(size_t)b * S + sbase + j] = logacc[j];
    }
}

// ---- kernel 3: softmax over S per batch; zero wref accumulator ----------
__global__ void softmax_kernel(const float* __restrict__ logits,
                               float* __restrict__ weights,
                               float* __restrict__ wref)
{
    __shared__ float red[256];
    const int b = blockIdx.x, t = threadIdx.x;
    const float* lrow = logits + (size_t)b * S;
    float lv[16];
    float m = -3.4e38f;
    #pragma unroll
    for (int i = 0; i < 16; ++i) { lv[i] = lrow[i * 256 + t]; m = fmaxf(m, lv[i]); }
    red[t] = m; __syncthreads();
    for (int s = 128; s > 0; s >>= 1) {
        if (t < s) red[t] = fmaxf(red[t], red[t + s]);
        __syncthreads();
    }
    const float mx = red[0];
    __syncthreads();
    float sum = 0.f;
    #pragma unroll
    for (int i = 0; i < 16; ++i) { lv[i] = __expf(lv[i] - mx); sum += lv[i]; }
    red[t] = sum; __syncthreads();
    for (int s = 128; s > 0; s >>= 1) {
        if (t < s) red[t] += red[t + s];
        __syncthreads();
    }
    const float inv = 1.f / red[0];
    float* wrow = weights + (size_t)b * S;
    #pragma unroll
    for (int i = 0; i < 16; ++i) wrow[i * 256 + t] = lv[i] * inv;
    wref[b * H + t] = 0.f;   // init accumulator for wsum_kernel
}

// ---- kernel 4: wref[b,h] = sum_s w[b,s] * ref[b,s,h] --------------------
__global__ void wsum_kernel(const float* __restrict__ ref,
                            const float* __restrict__ weights,
                            float* __restrict__ wref)
{
    const int b  = blockIdx.y;
    const int s0 = blockIdx.x * 256;
    const int h  = threadIdx.x;
    const float* rp = ref + ((size_t)b * S + s0) * H + h;
    const float* wp = weights + (size_t)b * S + s0;
    float acc = 0.f;
    for (int s = 0; s < 256; ++s) acc = __builtin_fmaf(wp[s], rp[(size_t)s * H], acc);
    atomicAdd(&wref[b * H + h], acc);
}

// ---- kernel 5: expected = wref @ Wr.T + br ------------------------------
__global__ void expected_kernel(const float* __restrict__ wref,
                                const float* __restrict__ Wr,
                                const float* __restrict__ br,
                                float* __restrict__ expected)
{
    __shared__ float x[H];
    const int b = blockIdx.x, h = threadIdx.x;
    x[h] = wref[b * H + h];
    __syncthreads();
    const float* wrow = Wr + (size_t)h * H;
    float acc = br[h];
    for (int j = 0; j < H; ++j) acc = __builtin_fmaf(wrow[j], x[j], acc);
    expected[b * H + h] = acc;
}

// ---- kernel 6: out = [output | expected] @ Wo.T + bo --------------------
__global__ void final_kernel(const float* __restrict__ outp,
                             const float* __restrict__ expected,
                             const float* __restrict__ Wo,
                             const float* __restrict__ bo,
                             float* __restrict__ res)
{
    __shared__ float cat[2 * H];
    const int b = blockIdx.x, h = threadIdx.x;
    cat[h]     = outp[b * H + h];
    cat[H + h] = expected[b * H + h];
    __syncthreads();
    const float* wrow = Wo + (size_t)h * (2 * H);
    float acc = bo[h];
    for (int j = 0; j < 2 * H; ++j) acc = __builtin_fmaf(wrow[j], cat[j], acc);
    res[b * H + h] = acc;
}

extern "C" void kernel_launch(void* const* d_in, const int* in_sizes, int n_in,
                              void* d_out, int out_size, void* d_ws, size_t ws_size,
                              hipStream_t stream)
{
    const float* outp  = (const float*)d_in[0];
    const float* query = (const float*)d_in[1];
    const float* ref   = (const float*)d_in[2];
    const float* Wq    = (const float*)d_in[3];
    const float* bq    = (const float*)d_in[4];
    const float* Wr    = (const float*)d_in[5];
    const float* br    = (const float*)d_in[6];
    const float* Wo    = (const float*)d_in[7];
    const float* bo    = (const float*)d_in[8];
    const float* V     = (const float*)d_in[9];
    float* res = (float*)d_out;

    // workspace layout
    char* ws = (char*)d_ws;
    float*          q        = (float*)(ws);                              // 64 KB
    unsigned short* WrBf     = (unsigned short*)(ws + 65536);             // 128 KB
    float*          logits   = (float*)(ws + 65536 + 131072);             // 1 MB
    float*          weights  = (float*)(ws + 65536 + 131072 + 1048576);   // 1 MB
    float*          wref     = (float*)(ws + 65536 + 131072 + 2097152);   // 64 KB
    float*          expected = (float*)(ws + 65536 + 131072 + 2097152 + 65536);

    wr_to_bf16_kernel<<<dim3(H * H / 256), dim3(256), 0, stream>>>(Wr, WrBf);
    qproj_kernel<<<dim3(BB), dim3(H), 0, stream>>>(query, Wq, bq, q);
    logits_kernel<<<dim3(S / STILE, BB), dim3(256), 0, stream>>>(ref, WrBf, q, br, V, logits);
    softmax_kernel<<<dim3(BB), dim3(256), 0, stream>>>(logits, weights, wref);
    wsum_kernel<<<dim3(S / 256, BB), dim3(256), 0, stream>>>(ref, weights, wref);
    expected_kernel<<<dim3(BB), dim3(H), 0, stream>>>(wref, Wr, br, expected);
    final_kernel<<<dim3(BB), dim3(H), 0, stream>>>(outp, expected, Wo, bo, res);
}
